// SeqRecModel_24060406792460
// MI455X (gfx1250) — compile-verified, matrix operand order checked by reading, not run on hardware
//
#include <hip/hip_runtime.h>
#include <math.h>

// ---------------- types ----------------
typedef __attribute__((ext_vector_type(16))) __bf16 v16bf;
typedef __attribute__((ext_vector_type(8)))  __bf16 v8bf;
typedef __attribute__((ext_vector_type(4)))  __bf16 v4bf;
typedef __attribute__((ext_vector_type(8)))  float  v8f;

// ---------------- model constants ----------------
constexpr int Bc  = 32;
constexpr int Lc  = 512;
constexpr int Dc  = 512;
constexpr int Hc  = 8;
constexpr int NLc = 2;
constexpr int DFFc = 2048;
constexpr int NBc = 32;
constexpr int Mc  = Bc * Lc;          // 16384 rows
#define MAX_DT_F 2592000.0f           // 3600*24*30

static __device__ inline v16bf mk16(v8bf lo, v8bf hi) {
  return __builtin_shufflevector(lo, hi, 0,1,2,3,4,5,6,7,8,9,10,11,12,13,14,15);
}

static __device__ inline v8f wmma_bf16(v16bf a, v16bf b, v8f c) {
  return __builtin_amdgcn_wmma_f32_16x16x32_bf16(false, a, false, b, (short)0, c, false, false);
}

// ---------------- elementwise kernels ----------------

__global__ __launch_bounds__(256) void cvt_bf16_kernel(const float* __restrict__ s,
                                                       __bf16* __restrict__ d, int n) {
  int i = blockIdx.x * 256 + threadIdx.x;
  if (i < n) d[i] = (__bf16)s[i];
}

__global__ __launch_bounds__(256) void embed_kernel(const int* __restrict__ seq,
                                                    const float* __restrict__ item_emb,
                                                    const float* __restrict__ pos_emb,
                                                    float* __restrict__ xf,
                                                    __bf16* __restrict__ xb) {
  size_t idx = (size_t)blockIdx.x * 256 + threadIdx.x;      // B*L*D threads exactly
  int d = (int)(idx & (Dc - 1));
  size_t t = idx >> 9;
  int l = (int)(t & (Lc - 1));
  int b = (int)(t >> 9);
  int it = seq[(size_t)b * Lc + l];
  float v = item_emb[(size_t)it * Dc + d] + pos_emb[(size_t)l * Dc + d];
  xf[idx] = v;
  xb[idx] = (__bf16)v;
}

__global__ __launch_bounds__(256) void bucket_kernel(const float* __restrict__ ts,
                                                     unsigned char* __restrict__ dt8) {
  size_t idx = (size_t)blockIdx.x * 256 + threadIdx.x;      // B*L*L threads exactly
  int k = (int)(idx & (Lc - 1));
  size_t t = idx >> 9;
  int q = (int)(t & (Lc - 1));
  int b = (int)(t >> 9);
  float delta = ts[(size_t)b * Lc + q] - ts[(size_t)b * Lc + k];
  float a = fabsf(delta);
  float ac = fmaxf(a, 1.0f);
  const float scale = log2f(MAX_DT_F) / (float)(NBc - 2);
  int bi = (int)(log2f(ac) / scale) + 1;                    // log2(ac) >= 0 -> trunc == floor
  bi = min(max(bi, 1), NBc - 2);
  if (ac <= 1.0f)     bi = 0;
  if (ac >= MAX_DT_F) bi = NBc - 1;
  dt8[idx] = (unsigned char)bi;
}

// wave-per-row residual-add + LayerNorm; writes f32 and bf16 copies
__global__ __launch_bounds__(256) void add_ln_kernel(const float* __restrict__ X,
                                                     const float* __restrict__ Dlt,
                                                     const float* __restrict__ g,
                                                     const float* __restrict__ be,
                                                     float* __restrict__ Of,
                                                     __bf16* __restrict__ Ob) {
  int row  = blockIdx.x * 8 + (threadIdx.x >> 5);
  int lane = threadIdx.x & 31;
  const float* xr = X   + (size_t)row * Dc;
  const float* dr = Dlt + (size_t)row * Dc;
  float v[16];
  float sum = 0.f, sq = 0.f;
#pragma unroll
  for (int j = 0; j < 16; ++j) {
    int d = j * 32 + lane;
    float t = xr[d] + dr[d];
    v[j] = t; sum += t; sq += t * t;
  }
#pragma unroll
  for (int off = 16; off; off >>= 1) {
    sum += __shfl_xor(sum, off, 32);
    sq  += __shfl_xor(sq,  off, 32);
  }
  float mean = sum * (1.0f / Dc);
  float var  = sq * (1.0f / Dc) - mean * mean;
  float rstd = rsqrtf(var + 1e-5f);
#pragma unroll
  for (int j = 0; j < 16; ++j) {
    int d = j * 32 + lane;
    float o = (v[j] - mean) * rstd * g[d] + be[d];
    Of[(size_t)row * Dc + d] = o;
    Ob[(size_t)row * Dc + d] = (__bf16)o;
  }
}

// ---------------- tiled WMMA GEMM ----------------
// C[M,N] = A[M,K](bf16) * B[K,N](bf16) + bias[N]; optional ReLU; store f32 or bf16.
// Block: 256 thr = 8 waves. Tile 64(M) x 128(N). Wave (wm,wn) -> rows wm*16, cols wn*64 (4 wmma).
template <bool RELU, bool OUTBF>
__global__ __launch_bounds__(256) void gemm_kernel(const __bf16* __restrict__ A,
                                                   const __bf16* __restrict__ Bw,
                                                   const float* __restrict__ bias,
                                                   float* __restrict__ Cf,
                                                   __bf16* __restrict__ Cb,
                                                   int M, int N, int K) {
  constexpr int LDA = 40;   // 32 + pad (halfs); row = 80B (16B aligned)
  __shared__ __align__(16) __bf16 sA[64 * LDA];
  __shared__ __align__(16) __bf16 sB[128 * LDA];   // transposed: [n][k]

  const int wid  = threadIdx.x >> 5;
  const int lane = threadIdx.x & 31;
  const int wm = wid & 3, wn = wid >> 2;
  const int m0 = blockIdx.y * 64, n0 = blockIdx.x * 128;
  const int nn = lane & 15;
  const int kb8 = (lane >> 4) << 3;

  v8f acc[4] = {};

  for (int k0 = 0; k0 < K; k0 += 32) {
    __syncthreads();
    // stage A tile 64x32 (one 16B load per thread)
    {
      int r = threadIdx.x >> 2, gcol = (threadIdx.x & 3) * 8;
      const v8bf* src = (const v8bf*)(A + (size_t)(m0 + r) * K + k0 + gcol);
      *(v8bf*)(&sA[r * LDA + gcol]) = *src;
    }
    // stage B tile 32x128 transposed into sB[n][k]: two 16B loads + scalar LDS scatter
    {
#pragma unroll
      for (int it = 0; it < 2; ++it) {
        int eid = threadIdx.x + it * 256;          // 0..511
        int kk = eid >> 4;                         // 0..31
        int nc = (eid & 15) * 8;                   // 0..120
        v8bf bvv = *(const v8bf*)(Bw + (size_t)(k0 + kk) * N + n0 + nc);
#pragma unroll
        for (int j = 0; j < 8; ++j) sB[(nc + j) * LDA + kk] = bvv[j];
      }
    }
    __syncthreads();

    const __bf16* ap = &sA[(wm * 16 + nn) * LDA];
    v16bf afrag = mk16(*(const v8bf*)(ap + kb8), *(const v8bf*)(ap + kb8 + 16));
#pragma unroll
    for (int nt = 0; nt < 4; ++nt) {
      const __bf16* bp = &sB[(wn * 64 + nt * 16 + nn) * LDA];
      v16bf bfrag = mk16(*(const v8bf*)(bp + kb8), *(const v8bf*)(bp + kb8 + 16));
      acc[nt] = wmma_bf16(afrag, bfrag, acc[nt]);
    }
  }

  const int hi = lane >> 4;
#pragma unroll
  for (int nt = 0; nt < 4; ++nt) {
    int col = n0 + wn * 64 + nt * 16 + nn;
    float bv = bias[col];
#pragma unroll
    for (int r = 0; r < 8; ++r) {
      int row = m0 + wm * 16 + r + 8 * hi;
      float v = acc[nt][r] + bv;
      if (RELU) v = fmaxf(v, 0.0f);
      if (OUTBF) Cb[(size_t)row * N + col] = (__bf16)v;
      else       Cf[(size_t)row * N + col] = v;
    }
  }
}

// ---------------- fused flash attention with time bias ----------------
// grid (L/64, H, B), block 128 = 4 waves; wave owns 16 q rows.
__global__ __launch_bounds__(128) void attn_kernel(const __bf16* __restrict__ Q,
                                                   const __bf16* __restrict__ Kb,
                                                   const __bf16* __restrict__ Vb,
                                                   const unsigned char* __restrict__ dt8,
                                                   const float* __restrict__ tbias_l,
                                                   __bf16* __restrict__ Ob) {
  constexpr int LDK = 72;   // 64 + pad (halfs)
  constexpr int LDP = 40;   // 32 + pad
  __shared__ __align__(16) __bf16 sK[32 * LDK];
  __shared__ __align__(16) __bf16 sVt[64 * LDP];   // [d][k]
  __shared__ __align__(16) __bf16 sP[4][16 * LDP];
  __shared__ __align__(16) unsigned char sDT[64 * 32];  // dt8 tile: [qloc][kloc]
  __shared__ float stb[NBc * Hc];

  const int qt = blockIdx.x, h = blockIdx.y, b = blockIdx.z;
  const int wid = threadIdx.x >> 5, lane = threadIdx.x & 31;
  const int qbase = qt * 64 + wid * 16;
  const int hi = lane >> 4, nn = lane & 15, kb8 = hi * 8;

  for (int i = threadIdx.x; i < NBc * Hc; i += 128) stb[i] = tbias_l[i];

  // Q fragments (A-matrix 16x32 layout), d-chunks [0,32) and [32,64)
  const __bf16* qrow = Q + ((size_t)(b * Lc + qbase + nn) * Dc + h * 64);
  v16bf qa0 = mk16(*(const v8bf*)(qrow + kb8),      *(const v8bf*)(qrow + kb8 + 16));
  v16bf qa1 = mk16(*(const v8bf*)(qrow + 32 + kb8), *(const v8bf*)(qrow + 32 + kb8 + 16));

  v8f o[4] = {};
  float mi[8], li[8];
#pragma unroll
  for (int r = 0; r < 8; ++r) { mi[r] = -3.0e38f; li[r] = 0.0f; }

  const unsigned char* dtb = dt8 + (size_t)b * Lc * Lc;
  const int nchunks = 2 * qt + 2;                   // causal: kbase <= qt*64+32

  for (int kt = 0; kt < nchunks; ++kt) {
    const int kbase = kt * 32;
    __syncthreads();
    // stage K chunk: 32 rows x 64 halfs (two 16B loads per thread)
#pragma unroll
    for (int it = 0; it < 2; ++it) {
      int cid = threadIdx.x + it * 128;             // 0..255
      int r = cid >> 3, c = (cid & 7) * 8;
      *(v8bf*)(&sK[r * LDK + c]) =
          *(const v8bf*)(Kb + ((size_t)(b * Lc + kbase + r) * Dc + h * 64 + c));
    }
    // stage V chunk transposed: four 8B loads + scalar LDS scatter
#pragma unroll
    for (int it = 0; it < 4; ++it) {
      int idx = threadIdx.x + it * 128;             // 0..511
      int r = idx >> 4, d0 = (idx & 15) * 4;
      v4bf vv = *(const v4bf*)(Vb + ((size_t)(b * Lc + kbase + r) * Dc + h * 64 + d0));
#pragma unroll
      for (int j = 0; j < 4; ++j) sVt[(d0 + j) * LDP + r] = vv[j];
    }
    // stage dt8 tile 64 q-rows x 32 k (one 16B load per thread)
    {
      int r = threadIdx.x >> 1, off = (threadIdx.x & 1) * 16;
      *(uint4*)(&sDT[r * 32 + off]) =
          *(const uint4*)(dtb + (size_t)(qt * 64 + r) * Lc + kbase + off);
    }
    __syncthreads();

    // S = Q K^T for two 16-key subtiles, accumulated over d (2 wmma each)
    v8f s[2];
#pragma unroll
    for (int ks = 0; ks < 2; ++ks) {
      const __bf16* kp = &sK[(ks * 16 + nn) * LDK];
      v16bf b0 = mk16(*(const v8bf*)(kp + kb8),      *(const v8bf*)(kp + kb8 + 16));
      v16bf b1 = mk16(*(const v8bf*)(kp + 32 + kb8), *(const v8bf*)(kp + 32 + kb8 + 16));
      v8f t = {};
      t = wmma_bf16(qa0, b0, t);
      t = wmma_bf16(qa1, b1, t);
      s[ks] = t;
    }

    // scale + time bias + causal mask + online softmax
    float alpha[8];
#pragma unroll
    for (int r = 0; r < 8; ++r) {
      int qloc = wid * 16 + r + 8 * hi;             // q row within block tile
      int q  = qt * 64 + qloc;
      int k0 = kbase + nn;
      int k1 = kbase + 16 + nn;
      float s0 = s[0][r] * 0.125f + stb[(int)sDT[qloc * 32 + nn] * Hc + h];
      float s1 = s[1][r] * 0.125f + stb[(int)sDT[qloc * 32 + 16 + nn] * Hc + h];
      if (k0 > q) s0 = -10000.0f;
      if (k1 > q) s1 = -10000.0f;
      float cm = fmaxf(s0, s1);
#pragma unroll
      for (int mm = 8; mm; mm >>= 1) cm = fmaxf(cm, __shfl_xor(cm, mm, 32));
      float nm = fmaxf(mi[r], cm);
      alpha[r] = __expf(mi[r] - nm);
      mi[r] = nm;
      float e0 = __expf(s0 - nm);
      float e1 = __expf(s1 - nm);
      float rs = e0 + e1;
#pragma unroll
      for (int mm = 8; mm; mm >>= 1) rs += __shfl_xor(rs, mm, 32);
      li[r] = li[r] * alpha[r] + rs;
      int mrow = r + 8 * hi;
      sP[wid][mrow * LDP + nn]      = (__bf16)e0;
      sP[wid][mrow * LDP + 16 + nn] = (__bf16)e1;
    }
    // rescale running output
#pragma unroll
    for (int nt = 0; nt < 4; ++nt)
#pragma unroll
      for (int r = 0; r < 8; ++r) o[nt][r] *= alpha[r];

    // O += P * V  (per-wave LDS round trip; same-wave DS ops are in-order)
    const __bf16* pp = &sP[wid][nn * LDP];
    v16bf pf = mk16(*(const v8bf*)(pp + kb8), *(const v8bf*)(pp + kb8 + 16));
#pragma unroll
    for (int nt = 0; nt < 4; ++nt) {
      const __bf16* vp = &sVt[(nt * 16 + nn) * LDP];
      v16bf vf = mk16(*(const v8bf*)(vp + kb8), *(const v8bf*)(vp + kb8 + 16));
      o[nt] = wmma_bf16(pf, vf, o[nt]);
    }
  }

  // normalize + store
#pragma unroll
  for (int nt = 0; nt < 4; ++nt)
#pragma unroll
    for (int r = 0; r < 8; ++r) {
      int q = qbase + r + 8 * hi;
      float v = o[nt][r] / li[r];
      Ob[(size_t)(b * Lc + q) * Dc + h * 64 + nt * 16 + nn] = (__bf16)v;
    }
}

// ---------------- launcher ----------------
extern "C" void kernel_launch(void* const* d_in, const int* in_sizes, int n_in,
                              void* d_out, int out_size, void* d_ws, size_t ws_size,
                              hipStream_t stream) {
  const int*   seq      = (const int*)  d_in[0];
  const float* ts       = (const float*)d_in[1];
  const float* item_emb = (const float*)d_in[2];
  const float* pos_emb  = (const float*)d_in[3];
  const float* Wq = (const float*)d_in[4];
  const float* bq = (const float*)d_in[5];
  const float* Wk = (const float*)d_in[6];
  const float* bk = (const float*)d_in[7];
  const float* Wv = (const float*)d_in[8];
  const float* bv = (const float*)d_in[9];
  const float* tbias = (const float*)d_in[10];
  const float* Wo = (const float*)d_in[11];
  const float* bo = (const float*)d_in[12];
  const float* g1  = (const float*)d_in[13];
  const float* be1 = (const float*)d_in[14];
  const float* W1  = (const float*)d_in[15];
  const float* bf1 = (const float*)d_in[16];
  const float* W2  = (const float*)d_in[17];
  const float* bf2 = (const float*)d_in[18];
  const float* g2  = (const float*)d_in[19];
  const float* be2 = (const float*)d_in[20];
  float* out = (float*)d_out;

  // workspace carve (all sizes multiples of 256)
  char* p = (char*)d_ws;
  auto take = [&](size_t bytes) -> char* {
    char* r = p; p += (bytes + 255) & ~(size_t)255; return r;
  };
  float*  xf  = (float*) take((size_t)Mc * Dc * 4);
  __bf16* xb  = (__bf16*)take((size_t)Mc * Dc * 2);
  __bf16* qb  = (__bf16*)take((size_t)Mc * Dc * 2);
  __bf16* kb  = (__bf16*)take((size_t)Mc * Dc * 2);
  __bf16* vb  = (__bf16*)take((size_t)Mc * Dc * 2);
  __bf16* ab  = (__bf16*)take((size_t)Mc * Dc * 2);
  float*  pf  = (float*) take((size_t)Mc * Dc * 4);
  __bf16* h1  = (__bf16*)take((size_t)Mc * DFFc * 2);
  unsigned char* dt8 = (unsigned char*)take((size_t)Bc * Lc * Lc);
  __bf16* wq = (__bf16*)take((size_t)NLc * Dc * Dc * 2);
  __bf16* wk = (__bf16*)take((size_t)NLc * Dc * Dc * 2);
  __bf16* wv = (__bf16*)take((size_t)NLc * Dc * Dc * 2);
  __bf16* wo = (__bf16*)take((size_t)NLc * Dc * Dc * 2);
  __bf16* w1 = (__bf16*)take((size_t)NLc * Dc * DFFc * 2);
  __bf16* w2 = (__bf16*)take((size_t)NLc * DFFc * Dc * 2);

  auto cvt = [&](const float* s, __bf16* d, size_t n) {
    cvt_bf16_kernel<<<(unsigned)((n + 255) / 256), 256, 0, stream>>>(s, d, (int)n);
  };
  cvt(Wq, wq, (size_t)NLc * Dc * Dc);
  cvt(Wk, wk, (size_t)NLc * Dc * Dc);
  cvt(Wv, wv, (size_t)NLc * Dc * Dc);
  cvt(Wo, wo, (size_t)NLc * Dc * Dc);
  cvt(W1, w1, (size_t)NLc * Dc * DFFc);
  cvt(W2, w2, (size_t)NLc * DFFc * Dc);

  embed_kernel<<<(Mc * Dc) / 256, 256, 0, stream>>>(seq, item_emb, pos_emb, xf, xb);
  bucket_kernel<<<(Bc * Lc * Lc) / 256, 256, 0, stream>>>(ts, dt8);

  const dim3 gD(Dc / 128, Mc / 64);       // N=512 GEMMs
  const dim3 gF(DFFc / 128, Mc / 64);     // N=2048 GEMM
  const dim3 gA(Lc / 64, Hc, Bc);

  for (int i = 0; i < NLc; ++i) {
    const size_t wD = (size_t)i * Dc * Dc;
    const size_t wF = (size_t)i * Dc * DFFc;
    // QKV projections (bf16 out)
    gemm_kernel<false, true><<<gD, 256, 0, stream>>>(xb, wq + wD, bq + i * Dc, nullptr, qb, Mc, Dc, Dc);
    gemm_kernel<false, true><<<gD, 256, 0, stream>>>(xb, wk + wD, bk + i * Dc, nullptr, kb, Mc, Dc, Dc);
    gemm_kernel<false, true><<<gD, 256, 0, stream>>>(xb, wv + wD, bv + i * Dc, nullptr, vb, Mc, Dc, Dc);
    // fused attention
    attn_kernel<<<gA, 128, 0, stream>>>(qb, kb, vb, dt8, tbias + (size_t)i * NBc * Hc, ab);
    // output projection (f32 out) + LN1
    gemm_kernel<false, false><<<gD, 256, 0, stream>>>(ab, wo + wD, bo + i * Dc, pf, nullptr, Mc, Dc, Dc);
    add_ln_kernel<<<Mc / 8, 256, 0, stream>>>(xf, pf, g1 + i * Dc, be1 + i * Dc, xf, xb);
    // FFN
    gemm_kernel<true, true><<<gF, 256, 0, stream>>>(xb, w1 + wF, bf1 + i * DFFc, nullptr, h1, Mc, DFFc, Dc);
    gemm_kernel<false, false><<<gD, 256, 0, stream>>>(h1, w2 + wF, bf2 + i * Dc, pf, nullptr, Mc, Dc, DFFc);
    float* lnout = (i == NLc - 1) ? out : xf;
    add_ln_kernel<<<Mc / 8, 256, 0, stream>>>(xf, pf, g2 + i * Dc, be2 + i * Dc, lnout, xb);
  }
}
